// AssociativeMemoryBank_17188459118912
// MI455X (gfx1250) — matrix operand order repaired and checked by reading, not compiled, not run on hardware
//
#include <hip/hip_runtime.h>
#include <hip/hip_bf16.h>

// ---------------------------------------------------------------------------
// AssociativeMemoryBank forward for MI455X (gfx1250), wave32 + WMMA bf16.
// Round 3: GLOBAL_LOAD_ASYNC_TO_LDS_B128 staging with corrected builtin
// signature (v4i addrspace(1)* / addrspace(3)*), per hipcc diagnostic.
// ---------------------------------------------------------------------------

#define B_    128
#define CAP_  65536
#define H_    512
#define NH_   8
#define HD_   64
#define GH_   128

typedef __attribute__((ext_vector_type(16))) __bf16 v16bf;
typedef __attribute__((ext_vector_type(8)))  __bf16 v8bf;
typedef __attribute__((ext_vector_type(8)))  float  v8f;
typedef __attribute__((ext_vector_type(4)))  int    v4i;

#if defined(__gfx1250__) && __has_builtin(__builtin_amdgcn_global_load_async_to_lds_b128)
#define HAVE_ASYNC_LDS 1
#else
#define HAVE_ASYNC_LDS 0
#endif

// ---------------- small helpers ----------------

__device__ __forceinline__ __bf16 f2bf(float f) {
  union { float f; unsigned u; } c; c.f = f;
  unsigned r = c.u + 0x7fffu + ((c.u >> 16) & 1u);   // round-to-nearest-even
  union { unsigned short s; __bf16 b; } o; o.s = (unsigned short)(r >> 16);
  return o.b;
}

__device__ __forceinline__ unsigned f2ord(float f) {  // monotonic float->uint
  union { float f; unsigned u; } c; c.f = f;
  return (c.u & 0x80000000u) ? ~c.u : (c.u | 0x80000000u);
}
__device__ __forceinline__ float ord2f(unsigned u) {
  union { float f; unsigned u; } c;
  c.u = (u & 0x80000000u) ? (u & 0x7fffffffu) : ~u;
  return c.f;
}

// Permuted position of k within a 32-wide A-tile row so that per-lane WMMA
// A-fragments ({0..7,16..23} / {8..15,24..31}) are contiguous 32B LDS loads.
// Operates on groups of 8 halves (= whole 16B chunks), so async B128 copies
// can honor the swizzle too.
__device__ __forceinline__ int apos(int kk) {
  int g = kk >> 3;                       // 0..3
  int ng = ((g & 1) << 1) | (g >> 1);    // 0->0, 1->2, 2->1, 3->3
  return (ng << 3) | (kk & 7);
}
__device__ __forceinline__ int agrp(int g) {  // group-of-8 swizzle
  return ((g & 1) << 1) | (g >> 1);
}

__device__ __forceinline__ v8f vzero() {
  v8f z;
#pragma unroll
  for (int i = 0; i < 8; i++) z[i] = 0.f;
  return z;
}

__device__ __forceinline__ v8f wmma_bf16(v16bf a, v16bf b, v8f c) {
  // (neg_a, A, neg_b, B, c_mod, C, reuse_a, reuse_b)
  return __builtin_amdgcn_wmma_f32_16x16x32_bf16(false, a, false, b,
                                                 (short)0, c, false, false);
}

// 16-byte global -> LDS copy: async DMA if available, else direct.
__device__ __forceinline__ void cp16(const __bf16* __restrict__ g,
                                     __bf16* __restrict__ l) {
#if HAVE_ASYNC_LDS
  __builtin_amdgcn_global_load_async_to_lds_b128(
      (__attribute__((address_space(1))) v4i*)(void*)g,
      (__attribute__((address_space(3))) v4i*)(void*)l, 0, 0);
#else
  *(v8bf*)l = *(const v8bf*)g;
#endif
}

// Drain this wave's outstanding async LDS copies (before the barrier).
__device__ __forceinline__ void wait_async() {
#if HAVE_ASYNC_LDS
#if __has_builtin(__builtin_amdgcn_s_wait_asynccnt)
  __builtin_amdgcn_s_wait_asynccnt(0);
#else
  asm volatile("s_wait_asynccnt 0x0" ::: "memory");
#endif
#endif
}

// ---------------- kernel 0: weight convert + accumulator init ----------------

__global__ __launch_bounds__(256) void prep_kernel(
    const float* __restrict__ in_proj_w, __bf16* __restrict__ Wb,
    unsigned* __restrict__ rowmax, float* __restrict__ Z,
    float* __restrict__ ctx, float* __restrict__ avgacc) {
  int i = blockIdx.x * 256 + threadIdx.x;
  if (i < 3 * H_ * H_) Wb[i] = f2bf(in_proj_w[i]);
  if (i < B_ * NH_) { rowmax[i] = 0u; Z[i] = 0.f; }
  if (i < B_ * H_)  { ctx[i] = 0.f; }
  if (i < CAP_)     { avgacc[i] = 0.f; }
}

// ---------------- kernel 1-3: projection GEMM  Out = A @ W^T + bias (bf16) ---
// A: rows x 512 f32, W: 512x512 row-major (W[j][i]) as bf16, Out: rows x 512 bf16
// Workgroup computes a 128x128 C tile; 8 waves as 4(M) x 2(N); each wave 32x64.

__global__ __launch_bounds__(256) void proj_gemm_bf16(
    const float* __restrict__ A, const __bf16* __restrict__ W,
    const float* __restrict__ bias, __bf16* __restrict__ Out) {
  __shared__ __align__(32) __bf16 Alds[128 * 32];
  __shared__ __align__(32) __bf16 Blds[128 * 32];

  const int tid = threadIdx.x;
  const int M0 = blockIdx.x * 128, N0 = blockIdx.y * 128;
  const int wave = tid >> 5, lane = tid & 31;
  const int wm = wave >> 1, wn = wave & 1;
  const int l16 = lane & 15, hsel = lane >> 4;

  v8f acc[2][4];
#pragma unroll
  for (int i = 0; i < 2; i++)
#pragma unroll
    for (int j = 0; j < 4; j++) acc[i][j] = vzero();

  for (int kb = 0; kb < H_; kb += 32) {
    // B tile (already bf16): 512 x 16B chunks, async copy to LDS
#pragma unroll
    for (int idx = tid; idx < 512; idx += 256) {
      int n = idx >> 2, g = idx & 3;
      cp16(&W[(size_t)(N0 + n) * H_ + kb + (g << 3)],
           &Blds[(n << 5) + (g << 3)]);
    }
    // A tile: f32 -> bf16 conversion, direct layout (VALU path)
    for (int idx = tid; idx < 128 * 32; idx += 256) {
      int m = idx >> 5, kk = idx & 31;
      Alds[(m << 5) + kk] = f2bf(A[(size_t)(M0 + m) * H_ + kb + kk]);
    }
    if (kb + 32 < H_)  // hint next A tile into cache (global_prefetch_b8)
      __builtin_prefetch(&A[(size_t)(M0 + (tid >> 1)) * H_ + kb + 32], 0, 0);
    wait_async();
    __syncthreads();

    v16bf af[2], bf[4];
#pragma unroll
    for (int mt = 0; mt < 2; mt++)
      af[mt] = *(const v16bf*)&Alds[((wm * 32 + mt * 16 + l16) << 5) + hsel * 16];
#pragma unroll
    for (int nt = 0; nt < 4; nt++)
      bf[nt] = *(const v16bf*)&Blds[((wn * 64 + nt * 16 + l16) << 5) + hsel * 16];
#pragma unroll
    for (int mt = 0; mt < 2; mt++)
#pragma unroll
      for (int nt = 0; nt < 4; nt++)
        acc[mt][nt] = wmma_bf16(af[mt], bf[nt], acc[mt][nt]);
    __syncthreads();
  }

#pragma unroll
  for (int mt = 0; mt < 2; mt++)
#pragma unroll
    for (int nt = 0; nt < 4; nt++) {
      int n = N0 + wn * 64 + nt * 16 + l16;
      float bv = bias[n];
#pragma unroll
      for (int r = 0; r < 8; r++) {
        int m = M0 + wm * 32 + mt * 16 + hsel * 8 + r;
        Out[(size_t)m * H_ + n] = f2bf(acc[mt][nt][r] + bv);
      }
    }
}

// ---------------- shared S-tile compute: S = Q_h (128x64) . K_h^T (64x128) ---
// Both tiles are pure bf16 copies -> fully async-staged.

__device__ __forceinline__ void score_tile(
    const __bf16* __restrict__ Qb, const __bf16* __restrict__ Kb,
    int head, int C0, __bf16* Alds, __bf16* Blds, v8f acc[2][4]) {
  const int tid = threadIdx.x;
  const int wave = tid >> 5, lane = tid & 31;
  const int wm = wave >> 1, wn = wave & 1;
  const int l16 = lane & 15, hsel = lane >> 4;
  const int col = head * HD_;

#pragma unroll
  for (int i = 0; i < 2; i++)
#pragma unroll
    for (int j = 0; j < 4; j++) acc[i][j] = vzero();

  for (int kb = 0; kb < HD_; kb += 32) {
    // A tile (Q): 512 x 16B chunks, direct layout
#pragma unroll
    for (int idx = tid; idx < 512; idx += 256) {
      int m = idx >> 2, g = idx & 3;
      cp16(&Qb[(size_t)m * H_ + col + kb + (g << 3)],
           &Alds[(m << 5) + (g << 3)]);
    }
    // B tile (K rows): 512 x 16B chunks, direct layout
#pragma unroll
    for (int idx = tid; idx < 512; idx += 256) {
      int n = idx >> 2, g = idx & 3;
      cp16(&Kb[(size_t)(C0 + n) * H_ + col + kb + (g << 3)],
           &Blds[(n << 5) + (g << 3)]);
    }
    wait_async();
    __syncthreads();

    v16bf af[2], bf[4];
#pragma unroll
    for (int mt = 0; mt < 2; mt++)
      af[mt] = *(const v16bf*)&Alds[((wm * 32 + mt * 16 + l16) << 5) + hsel * 16];
#pragma unroll
    for (int nt = 0; nt < 4; nt++)
      bf[nt] = *(const v16bf*)&Blds[((wn * 64 + nt * 16 + l16) << 5) + hsel * 16];
#pragma unroll
    for (int mt = 0; mt < 2; mt++)
#pragma unroll
      for (int nt = 0; nt < 4; nt++)
        acc[mt][nt] = wmma_bf16(af[mt], bf[nt], acc[mt][nt]);
    __syncthreads();
  }
}

// ---------------- kernel 4: per-row max of scores ----------------

__global__ __launch_bounds__(256) void score_rowmax_kernel(
    const __bf16* __restrict__ Qb, const __bf16* __restrict__ Kb,
    unsigned* __restrict__ rowmax) {
  __shared__ __align__(32) __bf16 Alds[128 * 32];
  __shared__ __align__(32) __bf16 Blds[128 * 32];
  v8f acc[2][4];
  const int C0 = blockIdx.x * 128, head = blockIdx.y;
  score_tile(Qb, Kb, head, C0, Alds, Blds, acc);

  const int tid = threadIdx.x, wave = tid >> 5, lane = tid & 31;
  const int wm = wave >> 1, l16 = lane & 15, hsel = lane >> 4;
#pragma unroll
  for (int mt = 0; mt < 2; mt++)
#pragma unroll
    for (int r = 0; r < 8; r++) {
      float v = fmaxf(fmaxf(acc[mt][0][r], acc[mt][1][r]),
                      fmaxf(acc[mt][2][r], acc[mt][3][r])) * 0.125f;
#pragma unroll
      for (int m = 1; m < 16; m <<= 1) v = fmaxf(v, __shfl_xor(v, m, 32));
      if (l16 == 0) {
        int row = head * 128 + wm * 32 + mt * 16 + hsel * 8 + r;
        atomicMax(&rowmax[row], f2ord(v));
      }
    }
}

// ---------------- kernel 5: per-row sum(exp(s-m)) ----------------

__global__ __launch_bounds__(256) void score_sumexp_kernel(
    const __bf16* __restrict__ Qb, const __bf16* __restrict__ Kb,
    const unsigned* __restrict__ rowmax, float* __restrict__ Z) {
  __shared__ __align__(32) __bf16 Alds[128 * 32];
  __shared__ __align__(32) __bf16 Blds[128 * 32];
  v8f acc[2][4];
  const int C0 = blockIdx.x * 128, head = blockIdx.y;
  score_tile(Qb, Kb, head, C0, Alds, Blds, acc);

  const int tid = threadIdx.x, wave = tid >> 5, lane = tid & 31;
  const int wm = wave >> 1, l16 = lane & 15, hsel = lane >> 4;
#pragma unroll
  for (int mt = 0; mt < 2; mt++)
#pragma unroll
    for (int r = 0; r < 8; r++) {
      int row = head * 128 + wm * 32 + mt * 16 + hsel * 8 + r;
      float mrow = ord2f(rowmax[row]);
      float s = 0.f;
#pragma unroll
      for (int nt = 0; nt < 4; nt++)
        s += __expf(acc[mt][nt][r] * 0.125f - mrow);
#pragma unroll
      for (int m = 1; m < 16; m <<= 1) s += __shfl_xor(s, m, 32);
      if (l16 == 0) atomicAdd(&Z[row], s);
    }
}

// ---------------- kernel 6: p = exp(s-m)/Z; avg_attn cols; ctx += P.V -------

__global__ __launch_bounds__(256) void attn_pass_kernel(
    const __bf16* __restrict__ Qb, const __bf16* __restrict__ Kb,
    const __bf16* __restrict__ Vb, const unsigned* __restrict__ rowmax,
    const float* __restrict__ Z, float* __restrict__ ctx,
    float* __restrict__ avgacc) {
  __shared__ __align__(32) __bf16 Alds[128 * 32];
  __shared__ __align__(32) __bf16 Blds[128 * 32];
  __shared__ __align__(32) __bf16 Plds[128 * 128];
  __shared__ __align__(32) __bf16 Vlds[64 * 32];
  v8f acc[2][4];
  const int C0 = blockIdx.x * 128, head = blockIdx.y;
  score_tile(Qb, Kb, head, C0, Alds, Blds, acc);

  const int tid = threadIdx.x, wave = tid >> 5, lane = tid & 31;
  const int wm = wave >> 1, wn = wave & 1, l16 = lane & 15, hsel = lane >> 4;

  float colsum[4] = {0.f, 0.f, 0.f, 0.f};
#pragma unroll
  for (int mt = 0; mt < 2; mt++)
#pragma unroll
    for (int r = 0; r < 8; r++) {
      int bl = wm * 32 + mt * 16 + hsel * 8 + r;
      int row = head * 128 + bl;
      float mrow = ord2f(rowmax[row]);
      float invz = 1.f / Z[row];
#pragma unroll
      for (int nt = 0; nt < 4; nt++) {
        float p = __expf(acc[mt][nt][r] * 0.125f - mrow) * invz;
        colsum[nt] += p;
        int cl = wn * 64 + nt * 16 + l16;
        Plds[bl * 128 + cl] = f2bf(p);
      }
    }
  // column sums -> avg_attn accumulator (mean over b,h applied later)
#pragma unroll
  for (int nt = 0; nt < 4; nt++) {
    float cs = colsum[nt] + __shfl_xor(colsum[nt], 16, 32);
    if (hsel == 0)
      atomicAdd(&avgacc[C0 + wn * 64 + nt * 16 + l16], cs);
  }

  // ctx partial: (128x128 P) @ (128x64 V_h), 4 K-chunks of 32
  v8f acc2[2][2];
#pragma unroll
  for (int i = 0; i < 2; i++)
#pragma unroll
    for (int j = 0; j < 2; j++) acc2[i][j] = vzero();

  for (int kb = 0; kb < 128; kb += 32) {
    // V tile is a transpose (stride-H source) -> element staging
    for (int idx = tid; idx < 64 * 32; idx += 256) {
      int d = idx >> 5, kk = idx & 31;
      Vlds[(d << 5) + kk] = Vb[(size_t)(C0 + kb + kk) * H_ + head * HD_ + d];
    }
    __syncthreads();   // also makes Plds visible on first iteration
    v16bf pf[2], vf[2];
#pragma unroll
    for (int mt = 0; mt < 2; mt++)
      pf[mt] = *(const v16bf*)&Plds[(wm * 32 + mt * 16 + l16) * 128 + kb + hsel * 16];
#pragma unroll
    for (int nt = 0; nt < 2; nt++)
      vf[nt] = *(const v16bf*)&Vlds[((wn * 32 + nt * 16 + l16) << 5) + hsel * 16];
#pragma unroll
    for (int mt = 0; mt < 2; mt++)
#pragma unroll
      for (int nt = 0; nt < 2; nt++)
        acc2[mt][nt] = wmma_bf16(pf[mt], vf[nt], acc2[mt][nt]);
    __syncthreads();
  }

#pragma unroll
  for (int mt = 0; mt < 2; mt++)
#pragma unroll
    for (int nt = 0; nt < 2; nt++)
#pragma unroll
      for (int r = 0; r < 8; r++) {
        int b = wm * 32 + mt * 16 + hsel * 8 + r;
        int d = wn * 32 + nt * 16 + l16;
        atomicAdd(&ctx[(size_t)b * H_ + head * HD_ + d], acc2[mt][nt][r]);
      }
}

// ---------------- kernel 7: out-proj + gate MLP + final memory_output -------

__global__ __launch_bounds__(256) void epilogue_kernel(
    const float* __restrict__ query, const float* __restrict__ ctx,
    const float* __restrict__ Wo, const float* __restrict__ bo,
    const float* __restrict__ gw1, const float* __restrict__ gb1,
    const float* __restrict__ gw2, const float* __restrict__ gb2,
    float* __restrict__ out0) {
  __shared__ float qrow[H_];
  __shared__ float crow[H_];
  __shared__ float g1[GH_];
  __shared__ float red[256];
  const int b = blockIdx.x, tid = threadIdx.x;
  for (int i = tid; i < H_; i += 256) {
    qrow[i] = query[(size_t)b * H_ + i];
    crow[i] = ctx[(size_t)b * H_ + i];
  }
  __syncthreads();
  if (tid < GH_) {
    float s = gb1[tid];
    for (int i = 0; i < H_; i++) s += qrow[i] * gw1[(size_t)tid * H_ + i];
    g1[tid] = fmaxf(s, 0.f);
  }
  __syncthreads();
  red[tid] = (tid < GH_) ? g1[tid] * gw2[tid] : 0.f;
  __syncthreads();
  for (int s = 128; s > 0; s >>= 1) {
    if (tid < s) red[tid] += red[tid + s];
    __syncthreads();
  }
  const float gate = 1.f / (1.f + __expf(-(red[0] + gb2[0])));
  for (int j = tid; j < H_; j += 256) {
    float s = bo[j];
    for (int i = 0; i < H_; i++) s += crow[i] * Wo[(size_t)j * H_ + i];
    out0[(size_t)b * H_ + j] = gate * s;
  }
}

// ---------------- kernel 8: usage / last_used update ----------------

__global__ __launch_bounds__(256) void usage_kernel(
    const float* __restrict__ usage, const float* __restrict__ last,
    const int* __restrict__ step, const float* __restrict__ avgacc,
    float* __restrict__ out_usage, float* __restrict__ out_last) {
  int c = blockIdx.x * 256 + threadIdx.x;
  if (c < CAP_) {
    float avg = avgacc[c] * (1.f / (float)(B_ * NH_));
    out_usage[c] = usage[c] + avg;
    out_last[c] = (avg > 0.001f) ? (float)step[0] : last[c];
  }
}

// ---------------- workspace layout ----------------

static constexpr size_t OFF_WB  = 0;                              // 3*H*H bf16
static constexpr size_t OFF_QB  = OFF_WB + 3ull * H_ * H_ * 2;    // B*H bf16
static constexpr size_t OFF_KB  = OFF_QB + (size_t)B_ * H_ * 2;   // CAP*H bf16
static constexpr size_t OFF_VB  = OFF_KB + (size_t)CAP_ * H_ * 2; // CAP*H bf16
static constexpr size_t OFF_RMX = OFF_VB + (size_t)CAP_ * H_ * 2; // B*NH u32
static constexpr size_t OFF_Z   = OFF_RMX + (size_t)B_ * NH_ * 4; // B*NH f32
static constexpr size_t OFF_CTX = OFF_Z + (size_t)B_ * NH_ * 4;   // B*H f32
static constexpr size_t OFF_AVG = OFF_CTX + (size_t)B_ * H_ * 4;  // CAP f32

extern "C" void kernel_launch(void* const* d_in, const int* in_sizes, int n_in,
                              void* d_out, int out_size, void* d_ws, size_t ws_size,
                              hipStream_t stream) {
  const float* query       = (const float*)d_in[0];
  const float* mem_keys    = (const float*)d_in[1];
  const float* mem_vals    = (const float*)d_in[2];
  const float* in_proj_w   = (const float*)d_in[3];
  const float* in_proj_b   = (const float*)d_in[4];
  const float* out_proj_w  = (const float*)d_in[5];
  const float* out_proj_b  = (const float*)d_in[6];
  const float* gate_w1     = (const float*)d_in[7];
  const float* gate_b1     = (const float*)d_in[8];
  const float* gate_w2     = (const float*)d_in[9];
  const float* gate_b2     = (const float*)d_in[10];
  const float* usage_count = (const float*)d_in[11];
  const float* last_used   = (const float*)d_in[12];
  const int*   step_ctr    = (const int*)d_in[13];

  char* ws = (char*)d_ws;
  __bf16*   Wb     = (__bf16*)(ws + OFF_WB);
  __bf16*   Qb     = (__bf16*)(ws + OFF_QB);
  __bf16*   Kb     = (__bf16*)(ws + OFF_KB);
  __bf16*   Vb     = (__bf16*)(ws + OFF_VB);
  unsigned* rowmax = (unsigned*)(ws + OFF_RMX);
  float*    Zd     = (float*)(ws + OFF_Z);
  float*    ctx    = (float*)(ws + OFF_CTX);
  float*    avgacc = (float*)(ws + OFF_AVG);

  float* out0      = (float*)d_out;              // memory_output (128x512)
  float* out_usage = out0 + (size_t)B_ * H_;     // new_usage (65536)
  float* out_last  = out_usage + (size_t)CAP_;   // new_last  (65536)

  // 0) convert weights + zero accumulators (re-run every call: deterministic)
  prep_kernel<<<(3 * H_ * H_ + 255) / 256, 256, 0, stream>>>(
      in_proj_w, Wb, rowmax, Zd, ctx, avgacc);

  // 1) Q / K / V projections (bf16 WMMA)
  proj_gemm_bf16<<<dim3(1, 4), 256, 0, stream>>>(
      query, Wb, in_proj_b, Qb);
  proj_gemm_bf16<<<dim3(CAP_ / 128, 4), 256, 0, stream>>>(
      mem_keys, Wb + (size_t)H_ * H_, in_proj_b + H_, Kb);
  proj_gemm_bf16<<<dim3(CAP_ / 128, 4), 256, 0, stream>>>(
      mem_vals, Wb + 2ull * H_ * H_, in_proj_b + 2 * H_, Vb);

  // 2) softmax statistics + attention pass (bf16 WMMA S-tiles, recomputed)
  dim3 sgrid(CAP_ / 128, NH_);
  score_rowmax_kernel<<<sgrid, 256, 0, stream>>>(Qb, Kb, rowmax);
  score_sumexp_kernel<<<sgrid, 256, 0, stream>>>(Qb, Kb, rowmax, Zd);
  attn_pass_kernel<<<sgrid, 256, 0, stream>>>(Qb, Kb, Vb, rowmax, Zd, ctx, avgacc);

  // 3) epilogues
  epilogue_kernel<<<B_, 256, 0, stream>>>(
      query, ctx, out_proj_w, out_proj_b, gate_w1, gate_b1, gate_w2, gate_b2, out0);
  usage_kernel<<<(CAP_ + 255) / 256, 256, 0, stream>>>(
      usage_count, last_used, step_ctr, avgacc, out_usage, out_last);
}
